// BatchMultiHeadGraphAttention_36515811951070
// MI455X (gfx1250) — compile-verified, matrix operand order checked by reading, not run on hardware
//
#include <hip/hip_runtime.h>
#include <hip/hip_bf16.h>
#include <stdint.h>

// Problem constants (from reference)
#define BS   4
#define NN   2048
#define FIN  128
#define NH   4
#define FOUT 64

typedef __attribute__((ext_vector_type(16))) _Float16 v16h;
typedef __attribute__((ext_vector_type(8)))  _Float16 v8h;
typedef __attribute__((ext_vector_type(8)))  float    v8f;
typedef __attribute__((ext_vector_type(4)))  float    v4f;

#define NEG_CAP (-3.0e38f)

// ---------------------------------------------------------------------------
// Kernel 0: transpose weights to f16  w_t[h][o][k] = (f16) w[h][k][o]
// ---------------------------------------------------------------------------
__global__ void prep_wt_kernel(const float* __restrict__ w, _Float16* __restrict__ w_t) {
    int idx = blockIdx.x * blockDim.x + threadIdx.x;
    if (idx < NH * FIN * FOUT) {
        int h = idx / (FIN * FOUT);
        int r = idx % (FIN * FOUT);
        int f = r / FOUT;
        int o = r % FOUT;
        w_t[((size_t)h * FOUT + o) * FIN + f] = (_Float16)w[idx];
    }
}

// ---------------------------------------------------------------------------
// Kernel 1: per (b,h,16-row tile): h = x @ w via WMMA f16->f32,
// store h transposed as f16 (h_t[bh][o][n]) and per-node attention scores.
// One wave (32 threads) per block.
// ---------------------------------------------------------------------------
__global__ __launch_bounds__(32)
void gat_gemm1_kernel(const float* __restrict__ x,
                      const _Float16* __restrict__ w_t,
                      const float* __restrict__ a_src,
                      const float* __restrict__ a_dst,
                      _Float16* __restrict__ h_t,
                      float* __restrict__ s_src,
                      float* __restrict__ s_dst) {
    const int lane = threadIdx.x;
    const int lh   = lane >> 4;     // lane half: 0 or 1
    const int mrow = lane & 15;
    const int tile = blockIdx.x % (NN / 16);
    const int bh   = blockIdx.x / (NN / 16);
    const int b    = bh >> 2;       // NH == 4
    const int head = bh & 3;
    const int i_base = tile * 16;

    v8f acc[4] = {};

    #pragma unroll
    for (int kc = 0; kc < FIN / 32; ++kc) {
        const int K0 = kc * 32;
        // A fragment: rows of x (f32 -> f16), K mapping e + 8*lh (+8 for e>=8)
        const float* xp = x + ((size_t)(b * NN + i_base + mrow)) * FIN + K0;
        v4f xa0 = *(const v4f*)(xp + 8 * lh);
        v4f xa1 = *(const v4f*)(xp + 8 * lh + 4);
        v4f xb0 = *(const v4f*)(xp + 16 + 8 * lh);
        v4f xb1 = *(const v4f*)(xp + 16 + 8 * lh + 4);
        v16h afrag;
        #pragma unroll
        for (int e = 0; e < 4; ++e) {
            afrag[e]      = (_Float16)xa0[e];
            afrag[4 + e]  = (_Float16)xa1[e];
            afrag[8 + e]  = (_Float16)xb0[e];
            afrag[12 + e] = (_Float16)xb1[e];
        }
        #pragma unroll
        for (int q = 0; q < 4; ++q) {
            const _Float16* wp = w_t + ((size_t)(head * FOUT + 16 * q + mrow)) * FIN + K0;
            v8h b0 = *(const v8h*)(wp + 8 * lh);
            v8h b1 = *(const v8h*)(wp + 16 + 8 * lh);
            v16h bfrag;
            #pragma unroll
            for (int e = 0; e < 8; ++e) { bfrag[e] = b0[e]; bfrag[8 + e] = b1[e]; }
            acc[q] = __builtin_amdgcn_wmma_f32_16x16x32_f16(
                false, afrag, false, bfrag, (short)0, acc[q], false, false);
        }
    }

    // Store h (f16, transposed) and accumulate tanh(h)·a_src / a_dst per row.
    float psrc[8], pdst[8];
    #pragma unroll
    for (int r = 0; r < 8; ++r) { psrc[r] = 0.0f; pdst[r] = 0.0f; }

    #pragma unroll
    for (int q = 0; q < 4; ++q) {
        const int col = 16 * q + mrow;
        const float av = a_src[head * FOUT + col];
        const float dv = a_dst[head * FOUT + col];
        #pragma unroll
        for (int r = 0; r < 8; ++r) {
            float hv = acc[q][r];
            h_t[((size_t)(bh * FOUT + col)) * NN + (i_base + r + 8 * lh)] = (_Float16)hv;
            float t = tanhf(hv);
            psrc[r] += t * av;
            pdst[r] += t * dv;
        }
    }
    // Reduce across the 16 lanes of each half (row = r + 8*lh)
    #pragma unroll
    for (int r = 0; r < 8; ++r) {
        float vs = psrc[r], vd = pdst[r];
        #pragma unroll
        for (int m = 1; m < 16; m <<= 1) {
            vs += __shfl_xor(vs, m, 32);
            vd += __shfl_xor(vd, m, 32);
        }
        if (mrow == r) {
            const int row = i_base + r + 8 * lh;
            s_src[(size_t)bh * NN + row] = vs;
            s_dst[(size_t)bh * NN + row] = vd;
        }
    }
}

// ---------------------------------------------------------------------------
// Kernel 2: fused flash-style GAT attention.
// One wave per (b,h,32-row tile) -> two 16-row sub-tiles share every
// B-fragment load (halves L2 traffic, 8 WMMAs per 32-col chunk).
// Online softmax; accumulator rescale skipped wave-uniformly when the
// running max is unchanged.
// ---------------------------------------------------------------------------
__global__ __launch_bounds__(32)
void gat_flash_kernel(const unsigned char* __restrict__ adj,
                      const _Float16* __restrict__ h_t,
                      const float* __restrict__ s_src,
                      const float* __restrict__ s_dst,
                      const float* __restrict__ bias,
                      float* __restrict__ out) {
    const int lane = threadIdx.x;
    const int lh   = lane >> 4;
    const int mrow = lane & 15;
    const int tile = blockIdx.x % (NN / 32);
    const int bh   = blockIdx.x / (NN / 32);
    const int b    = bh >> 2;
    const int head = bh & 3;
    const int i_base = tile * 32;
    const int row0   = i_base + mrow;        // sub-tile 0 score row
    const int row1   = i_base + 16 + mrow;   // sub-tile 1 score row

    const float ssrc0 = s_src[(size_t)bh * NN + row0];
    const float ssrc1 = s_src[(size_t)bh * NN + row1];
    const float* sdp  = s_dst + (size_t)bh * NN;
    const unsigned char* ap0 = adj + ((size_t)b * NN + row0) * NN;
    const unsigned char* ap1 = adj + ((size_t)b * NN + row1) * NN;

    v8f acc0[4] = {};
    v8f acc1[4] = {};
    float m0 = NEG_CAP, l0 = 0.0f;
    float m1 = NEG_CAP, l1 = 0.0f;

    for (int jc = 0; jc < NN / 32; ++jc) {
        const int j0 = jc * 32;

        // prefetch the adjacency stream (only true HBM stream) ~2 lines ahead
        __builtin_prefetch(ap0 + j0 + 256, 0, 1);
        __builtin_prefetch(ap1 + j0 + 256, 0, 1);

        // s_dst values in A-fragment K order (shared by both sub-tiles)
        v4f sd0 = *(const v4f*)(sdp + j0 + 8 * lh);
        v4f sd1 = *(const v4f*)(sdp + j0 + 8 * lh + 4);
        v4f sd2 = *(const v4f*)(sdp + j0 + 16 + 8 * lh);
        v4f sd3 = *(const v4f*)(sdp + j0 + 16 + 8 * lh + 4);
        // adjacency bytes (bool, 1 byte each), 8-byte aligned groups
        uint64_t a0m0 = *(const uint64_t*)(ap0 + j0 + 8 * lh);
        uint64_t a0m1 = *(const uint64_t*)(ap0 + j0 + 16 + 8 * lh);
        uint64_t a1m0 = *(const uint64_t*)(ap1 + j0 + 8 * lh);
        uint64_t a1m1 = *(const uint64_t*)(ap1 + j0 + 16 + 8 * lh);

        float sc0[16], sc1[16];
        #pragma unroll
        for (int e = 0; e < 16; ++e) {
            float sdv = (e < 4) ? sd0[e] : (e < 8) ? sd1[e - 4]
                      : (e < 12) ? sd2[e - 8] : sd3[e - 12];
            uint64_t k0 = (e < 8) ? ((a0m0 >> (8 * e)) & 0xFFull)
                                  : ((a0m1 >> (8 * (e - 8))) & 0xFFull);
            uint64_t k1 = (e < 8) ? ((a1m0 >> (8 * e)) & 0xFFull)
                                  : ((a1m1 >> (8 * (e - 8))) & 0xFFull);
            float v0 = ssrc0 + sdv;
            float v1 = ssrc1 + sdv;
            v0 = (v0 >= 0.0f) ? v0 : 0.2f * v0;   // leaky_relu(0.2)
            v1 = (v1 >= 0.0f) ? v1 : 0.2f * v1;
            sc0[e] = k0 ? v0 : NEG_CAP;
            sc1[e] = k1 ? v1 : NEG_CAP;
        }

        // chunk max per row (combine both lane halves)
        float lmax0 = sc0[0], lmax1 = sc1[0];
        #pragma unroll
        for (int e = 1; e < 16; ++e) {
            lmax0 = fmaxf(lmax0, sc0[e]);
            lmax1 = fmaxf(lmax1, sc1[e]);
        }
        lmax0 = fmaxf(lmax0, __shfl_xor(lmax0, 16, 32));
        lmax1 = fmaxf(lmax1, __shfl_xor(lmax1, 16, 32));

        const float mn0 = fmaxf(m0, lmax0);
        const float mn1 = fmaxf(m1, lmax1);

        // wave-uniform skip of the accumulator rescale when no row's max moved
        if (__ballot((mn0 > m0) || (mn1 > m1))) {
            const float scale0 = __expf(m0 - mn0);
            const float scale1 = __expf(m1 - mn1);
            float srow0[8], srow1[8];
            #pragma unroll
            for (int r = 0; r < 8; ++r) {
                srow0[r] = __shfl(scale0, r + 8 * lh, 32);
                srow1[r] = __shfl(scale1, r + 8 * lh, 32);
            }
            #pragma unroll
            for (int q = 0; q < 4; ++q)
                #pragma unroll
                for (int r = 0; r < 8; ++r) {
                    acc0[q][r] *= srow0[r];
                    acc1[q][r] *= srow1[r];
                }
            l0 *= scale0;
            l1 *= scale1;
            m0 = mn0;
            m1 = mn1;
        }

        float lsum0 = 0.0f, lsum1 = 0.0f;
        v16h afrag0, afrag1;
        #pragma unroll
        for (int e = 0; e < 16; ++e) {
            float p0 = __expf(sc0[e] - m0);
            float p1 = __expf(sc1[e] - m1);
            lsum0 += p0;
            lsum1 += p1;
            afrag0[e] = (_Float16)p0;
            afrag1[e] = (_Float16)p1;
        }
        lsum0 += __shfl_xor(lsum0, 16, 32);
        lsum1 += __shfl_xor(lsum1, 16, 32);
        l0 += lsum0;
        l1 += lsum1;

        // p(16x32) @ h(32x64): one B-fragment load feeds two WMMAs
        #pragma unroll
        for (int q = 0; q < 4; ++q) {
            const _Float16* hp = h_t + ((size_t)(bh * FOUT + 16 * q + mrow)) * NN + j0;
            v8h b0 = *(const v8h*)(hp + 8 * lh);
            v8h b1 = *(const v8h*)(hp + 16 + 8 * lh);
            v16h bfrag;
            #pragma unroll
            for (int e = 0; e < 8; ++e) { bfrag[e] = b0[e]; bfrag[8 + e] = b1[e]; }
            acc0[q] = __builtin_amdgcn_wmma_f32_16x16x32_f16(
                false, afrag0, false, bfrag, (short)0, acc0[q], false, false);
            acc1[q] = __builtin_amdgcn_wmma_f32_16x16x32_f16(
                false, afrag1, false, bfrag, (short)0, acc1[q], false, false);
        }
    }

    // epilogue: divide by softmax denominator, add bias, store [bs,H,n,FOUT] flat
    float linv0[8], linv1[8];
    #pragma unroll
    for (int r = 0; r < 8; ++r) {
        linv0[r] = 1.0f / __shfl(l0, r + 8 * lh, 32);
        linv1[r] = 1.0f / __shfl(l1, r + 8 * lh, 32);
    }
    #pragma unroll
    for (int q = 0; q < 4; ++q) {
        const int col = 16 * q + mrow;
        const float bq = bias[col];
        #pragma unroll
        for (int r = 0; r < 8; ++r) {
            const size_t base = ((size_t)(b * NH + head) * NN);
            out[(base + (i_base + r + 8 * lh)) * FOUT + col]      = acc0[q][r] * linv0[r] + bq;
            out[(base + (i_base + 16 + r + 8 * lh)) * FOUT + col] = acc1[q][r] * linv1[r] + bq;
        }
    }
}

// ---------------------------------------------------------------------------
extern "C" void kernel_launch(void* const* d_in, const int* in_sizes, int n_in,
                              void* d_out, int out_size, void* d_ws, size_t ws_size,
                              hipStream_t stream) {
    (void)in_sizes; (void)n_in; (void)out_size; (void)ws_size;
    const float*         x     = (const float*)d_in[0];
    const unsigned char* adj   = (const unsigned char*)d_in[1];   // jax bool = 1 byte
    const float*         w     = (const float*)d_in[2];
    const float*         a_src = (const float*)d_in[3];
    const float*         a_dst = (const float*)d_in[4];
    const float*         bias  = (const float*)d_in[5];
    float*               out   = (float*)d_out;

    // workspace carve (all 256B aligned)
    char* ws = (char*)d_ws;
    _Float16* w_t  = (_Float16*)ws;                               // H*64*128*2  = 64 KB
    _Float16* h_t  = (_Float16*)(ws + 65536);                     // 16*64*2048*2 = 4 MB
    float*    ssrc = (float*)(ws + 65536 + 4194304);              // 16*2048*4 = 128 KB
    float*    sdst = ssrc + (size_t)BS * NH * NN;

    const int wt_elems = NH * FIN * FOUT;
    prep_wt_kernel<<<(wt_elems + 255) / 256, 256, 0, stream>>>(w, w_t);

    const int tiles16 = BS * NH * (NN / 16);                      // 2048 waves
    gat_gemm1_kernel<<<tiles16, 32, 0, stream>>>(x, w_t, a_src, a_dst, h_t, ssrc, sdst);

    const int tiles32 = BS * NH * (NN / 32);                      // 1024 waves
    gat_flash_kernel<<<tiles32, 32, 0, stream>>>(adj, h_t, ssrc, sdst, bias, out);
}
